// OPNO2d_22539988369884
// MI455X (gfx1250) — compile-verified, matrix operand order checked by reading, not run on hardware
//
#include <hip/hip_runtime.h>
#include <math.h>

// ---------------------------------------------------------------------------
// WSWANO spectral neural operator for MI455X (gfx1250, wave32, WMMA).
// All GEMMs run on v_wmma_f32_16x16x32_f16 with fp32 accumulation.
// Weight stream (2.4 GB, single-use) is loaded non-temporal to preserve the
// L2-resident activations; activations/spectra stay regular-temporal.
// ---------------------------------------------------------------------------

typedef __attribute__((ext_vector_type(16))) _Float16 v16h;
typedef __attribute__((ext_vector_type(8)))  float    v8f;
typedef __attribute__((ext_vector_type(4)))  float    f32x4;

__device__ inline v8f zero8() {
  v8f z;
#pragma unroll
  for (int i = 0; i < 8; ++i) z[i] = 0.f;
  return z;
}

__device__ inline v8f wmma(v16h a, v16h b, v8f c) {
  // D = A(16x32 f16) * B(32x16 f16) + C(16x16 f32)
  return __builtin_amdgcn_wmma_f32_16x16x32_f16(
      /*neg_a=*/false, a, /*neg_b=*/false, b,
      /*c_mod=*/(short)0, c, /*reuse_a=*/false, /*reuse_b=*/false);
}

// Fragment load (A or B) from a row-major f16 array where the K dimension is
// contiguous. ISA 16-bit 16x32 layout: lane = (half<<4)|nm holds row nm,
// VGPR v packs k = ((v&3)<<1) + ((v>>2)<<4) + half*8 (+1 in high half).
__device__ inline v16h ld_frag(const _Float16* __restrict__ base, int row,
                               int stride, int kofs) {
  const _Float16* p = base + (size_t)row * stride + kofs;
  v16h f;
#pragma unroll
  for (int v = 0; v < 8; ++v) {
    int k0 = ((v & 3) << 1) + ((v >> 2) << 4);
    f[2 * v]     = p[k0];
    f[2 * v + 1] = p[k0 + 1];
  }
  return f;
}

__device__ inline v16h ld_frag_f32(const float* __restrict__ base, int row,
                                   int stride, int kofs) {
  const float* p = base + (size_t)row * stride + kofs;
  v16h f;
#pragma unroll
  for (int v = 0; v < 8; ++v) {
    int k0 = ((v & 3) << 1) + ((v >> 2) << 4);
    f[2 * v]     = (_Float16)p[k0];
    f[2 * v + 1] = (_Float16)p[k0 + 1];
  }
  return f;
}

__device__ inline float gelu_f(float x) {
  return 0.5f * x * (1.0f + erff(x * 0.70710678118654752f));
}

// ---------------------------------------------------------------------------
// S = sqrt(2/129) * sin(pi*(i+1)*(j+1)/129), symmetric, its own inverse.
// ---------------------------------------------------------------------------
__global__ __launch_bounds__(256) void k_gen_S(_Float16* __restrict__ Sh) {
  int idx = blockIdx.x * 256 + threadIdx.x;  // 16384
  int i = idx >> 7, j = idx & 127;
  float ang = 3.14159265358979323846f * (float)((i + 1) * (j + 1)) / 129.0f;
  Sh[idx] = (_Float16)(0.12451496f * sinf(ang));
}

// (B,H,W,3) -> xc (B,3,H,W) and channels 0..2 of x (B,64,H,W)
__global__ __launch_bounds__(256) void k_transpose_in(
    const float* __restrict__ xin, float* __restrict__ xc,
    float* __restrict__ xbuf) {
  int idx = blockIdx.x * 256 + threadIdx.x;  // 786432
  int c = idx % 3;
  int t = idx / 3;
  int w = t & 127; t >>= 7;
  int h = t & 127; int b = t >> 7;
  float v = xin[idx];
  xc[((size_t)(b * 3 + c) << 14) + h * 128 + w] = v;
  xbuf[((size_t)(b * 64 + c) << 14) + h * 128 + w] = v;
}

// ---------------------------------------------------------------------------
// Forward transform, cropped: acrop[0:66,0:66] = (S @ X @ S)[0:66,0:66].
// One workgroup per (b,c) image. dynsmem: Xh 32KB + ThT (80x128, T^t) 20KB.
// ---------------------------------------------------------------------------
__global__ __launch_bounds__(256) void k_fwd_dst(
    const float* __restrict__ src, const _Float16* __restrict__ Sh,
    float* __restrict__ acrop) {
  extern __shared__ unsigned char dynsmem[];
  _Float16* Xh  = (_Float16*)dynsmem;             // 128x128
  _Float16* ThT = (_Float16*)(dynsmem + 32768);   // [col 80][i 128]
  const int tid = threadIdx.x;
  const int wv = tid >> 5, lane = tid & 31;
  const int hlf = (lane >> 4) & 1, nm = lane & 15;
  const int bi = blockIdx.x;

  const float* ps = src + ((size_t)bi << 14);
  for (int t = tid; t < 4096; t += 256) {
    float4 v = ((const float4*)ps)[t];
    int o = t << 2;
    Xh[o] = (_Float16)v.x; Xh[o + 1] = (_Float16)v.y;
    Xh[o + 2] = (_Float16)v.z; Xh[o + 3] = (_Float16)v.w;
  }
  __syncthreads();

  // Phase 1: T = X @ S[:, :80]   (store transposed). S symmetric -> row reads.
  {
    const int mt = wv;  // 8 row-tiles over 8 waves
#pragma unroll
    for (int nt = 0; nt < 5; ++nt) {
      v8f acc = zero8();
#pragma unroll
      for (int ks = 0; ks < 4; ++ks) {
        v16h af = ld_frag(Xh, mt * 16 + nm, 128, ks * 32 + hlf * 8);
        v16h bf = ld_frag(Sh, nt * 16 + nm, 128, ks * 32 + hlf * 8);
        acc = wmma(af, bf, acc);
      }
      int col = nt * 16 + nm;
#pragma unroll
      for (int r = 0; r < 8; ++r)
        ThT[col * 128 + mt * 16 + r + hlf * 8] = (_Float16)acc[r];
    }
  }
  __syncthreads();

  // Phase 2: out = S[:80, :] @ T ; store rows/cols < 66 into acrop (pitch 72).
  for (int t = wv; t < 25; t += 8) {
    int mt = t / 5, nt = t % 5;
    v8f acc = zero8();
#pragma unroll
    for (int ks = 0; ks < 4; ++ks) {
      v16h af = ld_frag(Sh, mt * 16 + nm, 128, ks * 32 + hlf * 8);
      v16h bf = ld_frag(ThT, nt * 16 + nm, 128, ks * 32 + hlf * 8);
      acc = wmma(af, bf, acc);
    }
    int col = nt * 16 + nm;
    if (col < 66) {
#pragma unroll
      for (int r = 0; r < 8; ++r) {
        int rr = mt * 16 + r + hlf * 8;
        if (rr < 66) acrop[(size_t)bi * 4752 + rr * 72 + col] = acc[r];
      }
    }
  }
}

// ---------------------------------------------------------------------------
// Per-frequency patch matmul: out[b,o,x] = sum_i xp[b,i,x] * w[i,o,x].
// Grid: 128 frequency-tiles x 2 output-channel halves (256 WGs, ~96KB LDS ->
// 2 WGs co-resident per 320KB WGP for HBM saturation). Weights stream as
// full cachelines (nontemporal float4 over x) and are transposed to f16 in
// LDS so WMMA fragments are K-contiguous. Output is re-transposed through
// LDS (aliasing the weight buffer) for coalesced stores.
// ---------------------------------------------------------------------------
__global__ __launch_bounds__(256) void k_patch_mm(
    const float* __restrict__ acrop, const float* __restrict__ w,
    float* __restrict__ vbuf, int I, int Cin, int O) {
  extern __shared__ unsigned char dynsmem[];
  _Float16* wl = (_Float16*)dynsmem;                // 32o*32x*32k f16 = 64KB
  _Float16* al = (_Float16*)(dynsmem + 65536);      // 16b*32x*32k f16 = 32KB
  float*    st = (float*)dynsmem;                   // 32o*16b*32x f32 (alias)

  const int tid = threadIdx.x;
  const int wv = tid >> 5, lane = tid & 31;
  const int hlf = (lane >> 4) & 1, nm = lane & 15;
  const int x0 = (blockIdx.x >> 1) * 32;
  const int o0 = (blockIdx.x & 1) * 32;
  const int prow = x0 >> 6;     // output row p (x = p*64+q)
  const int q0 = x0 & 63;

  v8f acc[4][2];
#pragma unroll
  for (int a = 0; a < 4; ++a)
#pragma unroll
    for (int b = 0; b < 2; ++b) acc[a][b] = zero8();

  const int nk = (I + 31) >> 5;
  for (int kc = 0; kc < nk; ++kc) {
    const int i0 = kc << 5;
    __syncthreads();
    // Stream weight chunk: 32k x 32o x 32x fp32, nontemporal float4 reads.
    for (int tt = tid; tt < 8192; tt += 256) {
      int row = tt >> 3, seg = tt & 7;
      int kk = row >> 5, ol = row & 31;
      int i = i0 + kk, o = o0 + ol;
      f32x4 v = (f32x4)(0.f);
      if (i < I && o < O)
        v = __builtin_nontemporal_load(
            (const f32x4*)(w + (((size_t)(i * O + o)) << 12) + x0 + (seg << 2)));
      int xt = seg << 2;
      wl[(ol * 32 + xt + 0) * 32 + kk] = (_Float16)v.x;
      wl[(ol * 32 + xt + 1) * 32 + kk] = (_Float16)v.y;
      wl[(ol * 32 + xt + 2) * 32 + kk] = (_Float16)v.z;
      wl[(ol * 32 + xt + 3) * 32 + kk] = (_Float16)v.w;
    }
    // Gather activation chunk from cropped spectrum (shifted patch windows).
    for (int tt = tid; tt < 4096; tt += 256) {
      int row = tt >> 3, seg = tt & 7;
      int b = row >> 5, kk = row & 31;
      int i = i0 + kk;
      float v0 = 0.f, v1 = 0.f, v2 = 0.f, v3 = 0.f;
      if (i < I) {
        int c = i / 9, pp = i % 9, pi = pp / 3, pj = pp % 3;
        const float* s = acrop + (size_t)(b * Cin + c) * 4752 +
                         (pi + prow) * 72 + (pj + q0) + (seg << 2);
        v0 = s[0]; v1 = s[1]; v2 = s[2]; v3 = s[3];
      }
      int xt = seg << 2;
      al[(b * 32 + xt + 0) * 32 + kk] = (_Float16)v0;
      al[(b * 32 + xt + 1) * 32 + kk] = (_Float16)v1;
      al[(b * 32 + xt + 2) * 32 + kk] = (_Float16)v2;
      al[(b * 32 + xt + 3) * 32 + kk] = (_Float16)v3;
    }
    __syncthreads();
    const int xtb = wv << 2;  // each wave owns 4 frequencies
#pragma unroll
    for (int xi = 0; xi < 4; ++xi) {
      v16h af = ld_frag(al + (xtb + xi) * 32, nm, 1024, hlf * 8);
#pragma unroll
      for (int nt = 0; nt < 2; ++nt) {
        v16h bf = ld_frag(wl + (xtb + xi) * 32, nt * 16 + nm, 1024, hlf * 8);
        acc[xi][nt] = wmma(af, bf, acc[xi][nt]);
      }
    }
  }
  __syncthreads();
  // Transpose results through LDS for coalesced global stores.
  {
    const int xtb = wv << 2;
#pragma unroll
    for (int xi = 0; xi < 4; ++xi)
#pragma unroll
      for (int nt = 0; nt < 2; ++nt) {
        int ol = nt * 16 + nm;
#pragma unroll
        for (int r = 0; r < 8; ++r)
          st[(ol * 16 + r + hlf * 8) * 32 + xtb + xi] = acc[xi][nt][r];
      }
  }
  __syncthreads();
  for (int tt = tid; tt < 4096; tt += 256) {
    int row = tt >> 3, seg = tt & 7;
    int ol = row >> 4, b = row & 15;
    int o = o0 + ol;
    if (o < O) {
      float4 v = *(const float4*)(st + (ol * 16 + b) * 32 + seg * 4);
      *(float4*)(vbuf + (((size_t)(b * 64 + o)) << 12) + x0 + seg * 4) = v;
    }
  }
}

// ---------------------------------------------------------------------------
// Inverse transform from 64x64 support: out = S[:, :64] @ V @ S[:64, :].
// One workgroup per (b,o) image. Static LDS 24KB.
// ---------------------------------------------------------------------------
__global__ __launch_bounds__(256) void k_inv_dst(
    const float* __restrict__ vbuf, const _Float16* __restrict__ Sh,
    float* __restrict__ dst, int O, int cofs, int dogelu) {
  __shared__ _Float16 Vh[64 * 64];
  __shared__ _Float16 TT[128 * 64];  // [j][p] = T^t
  const int tid = threadIdx.x;
  const int wv = tid >> 5, lane = tid & 31;
  const int hlf = (lane >> 4) & 1, nm = lane & 15;
  const int b = blockIdx.x / O, o = blockIdx.x % O;

  const float* ps = vbuf + (((size_t)(b * 64 + o)) << 12);
  for (int t = tid; t < 1024; t += 256) {
    float4 v = ((const float4*)ps)[t];
    int idx = t << 2;
    Vh[idx] = (_Float16)v.x; Vh[idx + 1] = (_Float16)v.y;
    Vh[idx + 2] = (_Float16)v.z; Vh[idx + 3] = (_Float16)v.w;
  }
  __syncthreads();
  // Stage 1: T[p,j] = sum_{q<64} V[p,q] S[q,j]  (each wave: one j-tile)
  {
    const int jt = wv;
#pragma unroll
    for (int pt = 0; pt < 4; ++pt) {
      v8f acc = zero8();
#pragma unroll
      for (int ks = 0; ks < 2; ++ks) {
        v16h af = ld_frag(Vh, pt * 16 + nm, 64, ks * 32 + hlf * 8);
        v16h bf = ld_frag(Sh, jt * 16 + nm, 128, ks * 32 + hlf * 8);
        acc = wmma(af, bf, acc);
      }
      int j = jt * 16 + nm;
#pragma unroll
      for (int r = 0; r < 8; ++r)
        TT[j * 64 + pt * 16 + r + hlf * 8] = (_Float16)acc[r];
    }
  }
  __syncthreads();
  // Stage 2: out[i,j] = sum_{p<64} S[i,p] T[p,j]
  float* pd = dst + (((size_t)(b * 64 + cofs + o)) << 14);
  {
    const int it = wv;
#pragma unroll
    for (int jt = 0; jt < 8; ++jt) {
      v8f acc = zero8();
#pragma unroll
      for (int ks = 0; ks < 2; ++ks) {
        v16h af = ld_frag(Sh, it * 16 + nm, 128, ks * 32 + hlf * 8);
        v16h bf = ld_frag(TT, jt * 16 + nm, 64, ks * 32 + hlf * 8);
        acc = wmma(af, bf, acc);
      }
      int j = jt * 16 + nm;
#pragma unroll
      for (int r = 0; r < 8; ++r) {
        float v = acc[r];
        pd[(it * 16 + r + hlf * 8) * 128 + j] = dogelu ? gelu_f(v) : v;
      }
    }
  }
}

// ---------------------------------------------------------------------------
// Fused residual block: x = x + gelu(conv1x1(x) + wb + wsw), in place.
// One WG per (b, 256-pixel block). dynsmem: xh 32KB [px][c] + st 64KB [o][px].
// ---------------------------------------------------------------------------
__global__ __launch_bounds__(256) void k_conv_fuse(
    float* __restrict__ xbuf, const float* __restrict__ wsw,
    const float* __restrict__ ww, const float* __restrict__ wb) {
  extern __shared__ unsigned char dynsmem[];
  _Float16* xh = (_Float16*)dynsmem;            // [256 px][64 c]
  float*    st = (float*)(dynsmem + 32768);     // [64 o][256 px]
  const int tid = threadIdx.x;
  const int wv = tid >> 5, lane = tid & 31;
  const int hlf = (lane >> 4) & 1, nm = lane & 15;
  const int b = blockIdx.x >> 6, blk = blockIdx.x & 63;
  const int hw0 = blk << 8;
  const size_t base_b = (size_t)b << 20;

  for (int tt = tid; tt < 4096; tt += 256) {
    int c = tt >> 6, seg = tt & 63;
    float4 v = *(const float4*)(xbuf + base_b + ((size_t)c << 14) + hw0 + (seg << 2));
    int px = seg << 2;
    xh[(px + 0) * 64 + c] = (_Float16)v.x;
    xh[(px + 1) * 64 + c] = (_Float16)v.y;
    xh[(px + 2) * 64 + c] = (_Float16)v.z;
    xh[(px + 3) * 64 + c] = (_Float16)v.w;
  }
  __syncthreads();
#pragma unroll
  for (int mi = 0; mi < 2; ++mi) {
    int mt = wv * 2 + mi;
#pragma unroll
    for (int nt = 0; nt < 4; ++nt) {
      v8f acc = zero8();
#pragma unroll
      for (int ks = 0; ks < 2; ++ks) {
        v16h af = ld_frag(xh, mt * 16 + nm, 64, ks * 32 + hlf * 8);
        v16h bf = ld_frag_f32(ww, nt * 16 + nm, 64, ks * 32 + hlf * 8);
        acc = wmma(af, bf, acc);
      }
      int o = nt * 16 + nm;
#pragma unroll
      for (int r = 0; r < 8; ++r)
        st[o * 256 + mt * 16 + r + hlf * 8] = acc[r];
    }
  }
  __syncthreads();
  for (int idx = tid; idx < 16384; idx += 256) {
    int o = idx >> 8, px = idx & 255;
    size_t g = base_b + ((size_t)o << 14) + hw0 + px;
    float v = st[idx] + wb[o] + wsw[g];
    xbuf[g] = xbuf[g] + gelu_f(v);
  }
}

// ---------------------------------------------------------------------------
// Per-pixel MLP head: out = gelu(x@fc1+b1)@fc2+b2 (64 -> 128 -> 1).
// ---------------------------------------------------------------------------
__global__ __launch_bounds__(256) void k_head(
    const float* __restrict__ xbuf, const float* __restrict__ fc1,
    const float* __restrict__ b1, const float* __restrict__ fc2,
    const float* __restrict__ b2, float* __restrict__ out) {
  int px = blockIdx.x * 256 + threadIdx.x;  // 262144
  int b = px >> 14, hw = px & 16383;
  const float* ps = xbuf + ((size_t)b << 20) + hw;
  float xin[64];
#pragma unroll
  for (int c = 0; c < 64; ++c) xin[c] = ps[(size_t)c << 14];
  float acc = b2[0];
  for (int j = 0; j < 128; ++j) {
    float h = b1[j];
#pragma unroll
    for (int c = 0; c < 64; ++c) h += xin[c] * fc1[c * 128 + j];
    acc += gelu_f(h) * fc2[j];
  }
  out[px] = acc;
}

// ---------------------------------------------------------------------------
extern "C" void kernel_launch(void* const* d_in, const int* in_sizes, int n_in,
                              void* d_out, int out_size, void* d_ws,
                              size_t ws_size, hipStream_t stream) {
  (void)in_sizes; (void)n_in; (void)out_size; (void)ws_size;
  const float* x_in  = (const float*)d_in[0];
  const float* convl = (const float*)d_in[1];
  const float* conv_w[4] = {(const float*)d_in[2], (const float*)d_in[3],
                            (const float*)d_in[4], (const float*)d_in[5]};
  const float* w_w[4] = {(const float*)d_in[6], (const float*)d_in[8],
                         (const float*)d_in[10], (const float*)d_in[12]};
  const float* w_b[4] = {(const float*)d_in[7], (const float*)d_in[9],
                         (const float*)d_in[11], (const float*)d_in[13]};
  const float* fc1  = (const float*)d_in[14];
  const float* fc1b = (const float*)d_in[15];
  const float* fc2  = (const float*)d_in[16];
  const float* fc2b = (const float*)d_in[17];

  unsigned char* ws = (unsigned char*)d_ws;
  _Float16* Sh = (_Float16*)ws;  ws += 32768;
  float* xc    = (float*)ws;     ws += 3145728;    // (16,3,128,128)
  float* xbuf  = (float*)ws;     ws += 67108864;   // (16,64,128,128)
  float* acrop = (float*)ws;     ws += 19464192;   // (16*64, 66, 72)
  float* vbuf  = (float*)ws;     ws += 16777216;   // (16,64,64,64)
  float* wsw   = (float*)ws;     ws += 67108864;   // (16,64,128,128)

  k_gen_S<<<64, 256, 0, stream>>>(Sh);
  k_transpose_in<<<3072, 256, 0, stream>>>(x_in, xc, xbuf);

  // Lifting: x[:,3:] = gelu(wswano(xc, convl_w, 61))
  k_fwd_dst<<<16 * 3, 256, 53248, stream>>>(xc, Sh, acrop);
  k_patch_mm<<<256, 256, 98304, stream>>>(acrop, convl, vbuf, 27, 3, 61);
  k_inv_dst<<<16 * 61, 256, 0, stream>>>(vbuf, Sh, xbuf, 61, 3, 1);

  for (int l = 0; l < 4; ++l) {
    k_fwd_dst<<<16 * 64, 256, 53248, stream>>>(xbuf, Sh, acrop);
    k_patch_mm<<<256, 256, 98304, stream>>>(acrop, conv_w[l], vbuf, 576, 64, 64);
    k_inv_dst<<<16 * 64, 256, 0, stream>>>(vbuf, Sh, wsw, 64, 0, 0);
    k_conv_fuse<<<1024, 256, 98304, stream>>>(xbuf, wsw, w_w[l], w_b[l]);
  }

  k_head<<<1024, 256, 0, stream>>>(xbuf, fc1, fc1b, fc2, fc2b, (float*)d_out);
}
